// SharedSensorGRUEncoder_52664888983961
// MI455X (gfx1250) — compile-verified, matrix operand order checked by reading, not run on hardware
//
#include <hip/hip_runtime.h>

typedef _Float16 half16 __attribute__((ext_vector_type(16)));
typedef _Float16 half8  __attribute__((ext_vector_type(8)));
typedef float    f32x8  __attribute__((ext_vector_type(8)));
typedef float    f32x4  __attribute__((ext_vector_type(4)));

#define BB   16
#define LL   512
#define SS   256
#define HH   64
#define G3   192   // 3*H
#define IND  10

// fast sigmoid / tanh via v_exp_f32 + v_rcp_f32 (TRANS ops co-execute with WMMA)
__device__ __forceinline__ float fsigmoid(float x) {
  float e = __builtin_amdgcn_exp2f(-1.44269504f * x);
  return __builtin_amdgcn_rcpf(1.0f + e);
}
__device__ __forceinline__ float ftanh(float x) {
  float e = __builtin_amdgcn_exp2f(-2.88539008f * x);  // exp(-2x)
  return (1.0f - e) * __builtin_amdgcn_rcpf(1.0f + e);
}

// ---------------------------------------------------------------------------
// Kernel A: tproj[b][t][g] = sum_i tfeat[b][t][i] * w_ih[g][1+i] + b_ih[g]
//                            (+ b_hh[g] folded in for the r/z gates, g < 128)
// ---------------------------------------------------------------------------
__global__ void tproj_kernel(const float* __restrict__ tfeat,
                             const float* __restrict__ w_ih,
                             const float* __restrict__ b_ih,
                             const float* __restrict__ b_hh,
                             float* __restrict__ tp) {
  int idx = blockIdx.x * blockDim.x + threadIdx.x;
  if (idx >= BB * LL * G3) return;
  int g  = idx % G3;
  int bt = idx / G3;                       // b*L + t
  const float* tf = tfeat + (size_t)bt * 9;
  const float* w  = w_ih + (size_t)g * IND + 1;
  float acc = b_ih[g] + (g < 2 * HH ? b_hh[g] : 0.0f);
#pragma unroll
  for (int i = 0; i < 9; ++i) acc += tf[i] * w[i];
  tp[idx] = acc;
}

// ---------------------------------------------------------------------------
// Kernel B: the sequential GRU scan. One wave (one block) owns 16 sequences.
// Per step: gh(16x192) = h(16x64) @ w_hh^T via 24x v_wmma_f32_16x16x32_f16,
// with the r/z x-projection terms pre-loaded into the WMMA C operand.
// ---------------------------------------------------------------------------
__global__ void __launch_bounds__(32)
gru_scan(const float* __restrict__ r,
         const float* __restrict__ w_ih,
         const float* __restrict__ w_hh,
         const float* __restrict__ b_hh,
         const float* __restrict__ tp,
         float* __restrict__ out) {
  __shared__ _Float16 hbuf[16][HH];      // per-wave h tile, row-major [M][k], f16

  const int lane = threadIdx.x & 31;
  const int col  = lane & 15;            // N / M-row selector
  const int khi  = (lane >> 4) & 1;      // lane group
  const int koff = khi * 8;              // A/B fragment K sub-offset
  const int moff = khi * 8;              // D-layout M offset

  const int seq0 = blockIdx.x * 16;      // 16 consecutive sequences
  const int b    = seq0 >> 8;            // all 16 share one batch element
  const int s0   = seq0 & 255;

  // ---- preload w_hh as WMMA B fragments (f16) -> stays in VGPRs all 512 steps
  half16 bfrag[12][2];
#pragma unroll
  for (int q = 0; q < 12; ++q) {
    const int g = q * 16 + col;                  // output-gate column
#pragma unroll
    for (int c = 0; c < 2; ++c) {
      const float* wrow = w_hh + (size_t)g * HH + c * 32 + koff;
      f32x4 a0 = *(const f32x4*)(wrow);
      f32x4 a1 = *(const f32x4*)(wrow + 4);
      f32x4 a2 = *(const f32x4*)(wrow + 16);
      f32x4 a3 = *(const f32x4*)(wrow + 20);
      half16 f;
#pragma unroll
      for (int i = 0; i < 4; ++i) {
        f[i]      = (_Float16)a0[i];
        f[4 + i]  = (_Float16)a1[i];
        f[8 + i]  = (_Float16)a2[i];
        f[12 + i] = (_Float16)a3[i];
      }
      bfrag[q][c] = f;
    }
  }

  // per-lane constants: w_ih[:,0] column and b_hh for the n-gate tiles
  float w0[12];
#pragma unroll
  for (int q = 0; q < 12; ++q) w0[q] = w_ih[(size_t)(q * 16 + col) * IND];
  float bhhn[4];
#pragma unroll
  for (int q = 0; q < 4; ++q) bhhn[q] = b_hh[128 + q * 16 + col];

  // ---- zero hidden state (registers, f32 D-layout) and the LDS h tile
  f32x8 h_state[4];
#pragma unroll
  for (int q = 0; q < 4; ++q) h_state[q] = (f32x8){0,0,0,0,0,0,0,0};
  {
    uint4 z = {0u, 0u, 0u, 0u};
    uint4* hb = (uint4*)&hbuf[0][0];             // 2 KB = 128 x uint4
#pragma unroll
    for (int i = 0; i < 4; ++i) hb[lane + 32 * i] = z;
  }

  const float* rbase  = r + ((size_t)b * LL) * SS + s0;  // stride SS per step
  const float* tpbase = tp + ((size_t)b * LL) * G3;

  for (int t = 0; t < LL; ++t) {
    // keep next rows warm in L0/L2 (global_prefetch_b8)
    __builtin_prefetch(rbase + (size_t)(t + 8) * SS + col, 0, 0);
    __builtin_prefetch(tpbase + (size_t)(t + 8) * G3 + col * 12, 0, 0);

    // per-sequence input values for this step: lane reads the 8 it needs
    const float* rstep = rbase + (size_t)t * SS + moff;
    f32x4 rv0 = *(const f32x4*)(rstep);
    f32x4 rv1 = *(const f32x4*)(rstep + 4);

    // A fragments of h from LDS (ISA 16-bit A 16x32 layout: two 16B runs/lane)
    const _Float16* hrow = &hbuf[col][0];
    half16 afrag[2];
#pragma unroll
    for (int c = 0; c < 2; ++c) {
      half8 p0 = *(const half8*)(hrow + c * 32 + koff);
      half8 p1 = *(const half8*)(hrow + c * 32 + 16 + koff);
      half16 a;
#pragma unroll
      for (int i = 0; i < 8; ++i) { a[i] = p0[i]; a[8 + i] = p1[i]; }
      afrag[c] = a;
    }

    const float* tpt = tpbase + (size_t)t * G3;

#pragma unroll
    for (int q = 0; q < 4; ++q) {
      float tpr = tpt[q * 16 + col];          // includes b_ih + b_hh (r gate)
      float tpz = tpt[64 + q * 16 + col];     // includes b_ih + b_hh (z gate)
      float tpn = tpt[128 + q * 16 + col];    // includes b_ih only   (n gate)

      // fold the r/z-gate x-projection into the WMMA C operand
      f32x8 cr, cz;
#pragma unroll
      for (int rr = 0; rr < 8; ++rr) {
        float rv = (rr < 4) ? rv0[rr] : rv1[rr - 4];
        cr[rr] = tpr + rv * w0[q];
        cz[rr] = tpz + rv * w0[q + 4];
      }

      f32x8 zero = (f32x8){0,0,0,0,0,0,0,0};
      f32x8 dr = __builtin_amdgcn_wmma_f32_16x16x32_f16(false, afrag[0], false, bfrag[q][0],     (short)0, cr,   false, false);
      dr       = __builtin_amdgcn_wmma_f32_16x16x32_f16(false, afrag[1], false, bfrag[q][1],     (short)0, dr,   false, false);
      f32x8 dz = __builtin_amdgcn_wmma_f32_16x16x32_f16(false, afrag[0], false, bfrag[q + 4][0], (short)0, cz,   false, false);
      dz       = __builtin_amdgcn_wmma_f32_16x16x32_f16(false, afrag[1], false, bfrag[q + 4][1], (short)0, dz,   false, false);
      f32x8 dn = __builtin_amdgcn_wmma_f32_16x16x32_f16(false, afrag[0], false, bfrag[q + 8][0], (short)0, zero, false, false);
      dn       = __builtin_amdgcn_wmma_f32_16x16x32_f16(false, afrag[1], false, bfrag[q + 8][1], (short)0, dn,   false, false);

#pragma unroll
      for (int rr = 0; rr < 8; ++rr) {
        float rv = (rr < 4) ? rv0[rr] : rv1[rr - 4];
        float rg = fsigmoid(dr[rr]);
        float zg = fsigmoid(dz[rr]);
        // xn + rg*(dn + b_hh_n)  ->  fma chain
        float xn = tpn + rv * w0[q + 8];
        float ng = ftanh(__builtin_fmaf(rg, dn[rr] + bhhn[q], xn));
        float hv = ng + zg * (h_state[q][rr] - ng);
        h_state[q][rr] = hv;
        // write back f16 h for next step's A fragment (D layout -> row major)
        hbuf[rr + moff][q * 16 + col] = (_Float16)hv;
      }
    }
  }

  // ---- final h -> out[b][s][j]
#pragma unroll
  for (int q = 0; q < 4; ++q)
#pragma unroll
    for (int rr = 0; rr < 8; ++rr)
      out[((size_t)(seq0 + rr + moff)) * HH + q * 16 + col] = h_state[q][rr];
}

extern "C" void kernel_launch(void* const* d_in, const int* in_sizes, int n_in,
                              void* d_out, int out_size, void* d_ws, size_t ws_size,
                              hipStream_t stream) {
  const float* r     = (const float*)d_in[0];
  const float* tfeat = (const float*)d_in[1];
  const float* w_ih  = (const float*)d_in[2];
  const float* w_hh  = (const float*)d_in[3];
  const float* b_ih  = (const float*)d_in[4];
  const float* b_hh  = (const float*)d_in[5];
  float* out = (float*)d_out;
  float* tpw = (float*)d_ws;   // 16*512*192 f32 = 6.29 MB (L2-resident)

  const int ntp = BB * LL * G3;
  tproj_kernel<<<(ntp + 255) / 256, 256, 0, stream>>>(tfeat, w_ih, b_ih, b_hh, tpw);
  gru_scan<<<256, 32, 0, stream>>>(r, w_ih, w_hh, b_hh, tpw, out);
}
